// TaskDrivenSTGCNClassifier_48541720379460
// MI455X (gfx1250) — matrix operand order repaired and checked by reading, not compiled, not run on hardware
//
#include <hip/hip_runtime.h>
#include <hip/hip_bf16.h>

// ST-GCN forward for MI455X (gfx1250).
// All GEMM-like ops (GCN feature transform, graph GEMM, 1xK temporal conv as 9 tap-GEMMs,
// residual 1x1) run on v_wmma_f32_16x16x32_f16 with f32 accumulation.
// Each wave computes TWO stacked 16x16 output tiles sharing one B fragment (2x M-blocking):
// halves activation-operand loads, 2 WMMAs per 6 b128 loads.
// Layouts give aligned 16B vector loads for every fragment:
//   activations: channels-last  [B][N][T][Cpad]  (K = channels contiguous)
//   Z (feat out): nodes-last    [B][C][T][64]    (K = nodes contiguous for graph GEMM)

typedef _Float16 f16;
typedef __attribute__((ext_vector_type(16))) _Float16 v16h;
typedef __attribute__((ext_vector_type(8)))  _Float16 v8h;
typedef __attribute__((ext_vector_type(8)))  float    v8f;

#define NODES  62
#define TFULL  1024
#define THALF  512
#define BATCH  32
#define SLICES 64   // partial-stat slices per (b) for instance norm

// ---------------- WMMA fragment loaders (CDNA5 ISA 7.12.2 layouts) ----------------

// A-matrix 16x32 f16 from row-major [Mpad][Kpad] (Kpad % 32 == 0, 16B-aligned rows).
// lane L: row = L&15; lo lanes hold K {0..7,16..23}, hi lanes K {8..15,24..31}.
__device__ __forceinline__ v16h load_fragA(const f16* __restrict__ A, int m0, int Kpad, int k0) {
  int lane = threadIdx.x & 31;
  int row  = lane & 15;
  int kb   = (lane >> 4) * 8;
  const f16* base = A + (size_t)(m0 + row) * Kpad + k0;
  v8h a0 = *(const v8h*)(base + kb);        // K = kb..kb+7
  v8h a1 = *(const v8h*)(base + 16 + kb);   // K = 16+kb..16+kb+7
  return __builtin_shufflevector(a0, a1, 0, 1, 2, 3, 4, 5, 6, 7,
                                         8, 9, 10, 11, 12, 13, 14, 15);
}

// B-matrix 32x16 f16: K contiguous in memory. bk points at this lane's (col, k0) element.
// lo lanes: K {0..7,16..23}, hi lanes: K {8..15,24..31} -> two aligned 16B loads.
__device__ __forceinline__ v16h load_fragB(const f16* __restrict__ bk) {
  int kb = ((threadIdx.x & 31) >> 4) * 8;
  v8h lo = *(const v8h*)(bk + kb);
  v8h hi = *(const v8h*)(bk + 16 + kb);
  return __builtin_shufflevector(lo, hi, 0, 1, 2, 3, 4, 5, 6, 7,
                                         8, 9, 10, 11, 12, 13, 14, 15);
}

__device__ __forceinline__ v16h zero_frag() {
  v16h z;
#pragma unroll
  for (int j = 0; j < 16; ++j) z[j] = (f16)0.f;
  return z;
}

// ---------------- Feature GEMM: Z[b,co,t,n] = sum_ci W[ci,co] * act[b,n,t,ci] ----------------
// grid.y covers Cout/32; each wave does channel tiles [mt*32, mt*32+16) and [mt*32+16, mt*32+32).
__global__ void wmma_feat_kernel(const f16* __restrict__ act, int CS,
                                 const f16* __restrict__ wpack,
                                 f16* __restrict__ Z, int Cin, int Cout, int T) {
  int tTiles = T >> 4;
  int tile = blockIdx.x % tTiles;
  int rem  = blockIdx.x / tTiles;
  int n    = rem % NODES;
  int b    = rem / NODES;
  int m0   = blockIdx.y * 32;
  int lane = threadIdx.x & 31;
  int col  = lane & 15;
  int hi   = lane >> 4;
  int Kpad = (Cin + 31) & ~31;
  int t0   = tile << 4;

  v8f acc0, acc1;
#pragma unroll
  for (int r = 0; r < 8; ++r) { acc0[r] = 0.f; acc1[r] = 0.f; }

  const f16* bbase = act + (((size_t)(b * NODES + n)) * T + t0 + col) * CS;
  for (int kc = 0; kc < Cin; kc += 32) {
    v16h bfrag = load_fragB(bbase + kc);
    v16h af0 = load_fragA(wpack, m0,      Kpad, kc);
    v16h af1 = load_fragA(wpack, m0 + 16, Kpad, kc);
    acc0 = __builtin_amdgcn_wmma_f32_16x16x32_f16(false, af0, false, bfrag,
                                                  (short)0, acc0, false, false);
    acc1 = __builtin_amdgcn_wmma_f32_16x16x32_f16(false, af1, false, bfrag,
                                                  (short)0, acc1, false, false);
  }
  // Z nodes-last: [B][Cout][T][64]
  size_t zb = ((size_t)b * Cout + m0 + hi * 8) * T;
#pragma unroll
  for (int r = 0; r < 8; ++r) {
    Z[(zb + (size_t)r * T + t0 + col) * 64 + n]            = (f16)acc0[r];
    Z[(zb + (size_t)(r + 16) * T + t0 + col) * 64 + n]     = (f16)acc1[r];
  }
}

// ------------- Graph GEMM: Y[b,m,t,c] = bg[c] + sum_n Anorm[m,n] Z[b,c,t,n] -------------
// grid.y = 2; each wave does node tiles [mt*32, +16) and [mt*32+16, +32). An zero-padded in K.
__global__ void wmma_graph_kernel(const f16* __restrict__ Z, const f16* __restrict__ An,
                                  const float* __restrict__ bias,
                                  f16* __restrict__ Y, int C, int T) {
  int tTiles = T >> 4;
  int tile = blockIdx.x % tTiles;
  int rem  = blockIdx.x / tTiles;
  int c    = rem % C;
  int b    = rem / C;
  int m0   = blockIdx.y * 32;     // node tile base
  int lane = threadIdx.x & 31;
  int col  = lane & 15;
  int hi   = lane >> 4;
  int t0   = tile << 4;

  float bv = bias[c];
  v8f acc0, acc1;
#pragma unroll
  for (int r = 0; r < 8; ++r) { acc0[r] = bv; acc1[r] = bv; }

  const f16* bbase = Z + (((size_t)(b * C + c)) * T + t0 + col) * 64;
  for (int kc = 0; kc < 64; kc += 32) {
    v16h bfrag = load_fragB(bbase + kc);
    v16h af0 = load_fragA(An, m0,      64, kc);
    v16h af1 = load_fragA(An, m0 + 16, 64, kc);
    acc0 = __builtin_amdgcn_wmma_f32_16x16x32_f16(false, af0, false, bfrag,
                                                  (short)0, acc0, false, false);
    acc1 = __builtin_amdgcn_wmma_f32_16x16x32_f16(false, af1, false, bfrag,
                                                  (short)0, acc1, false, false);
  }
  // Y channels-last: [B][62][T][C]
#pragma unroll
  for (int r = 0; r < 8; ++r) {
    int ma = m0 + hi * 8 + r;
    int mb = ma + 16;
    if (ma < NODES)
      Y[(((size_t)b * NODES + ma) * T + t0 + col) * C + c] = (f16)acc0[r];
    if (mb < NODES)
      Y[(((size_t)b * NODES + mb) * T + t0 + col) * C + c] = (f16)acc1[r];
  }
}

// ------------- Temporal / residual conv (implicit GEMM over taps) -------------
// out[b,n,t',co] = bias[co] + sum_{tap,ci} w[tap][co][ci] * act[b,n, t'*stride+tap-pad, ci]
// grid.y covers Cout/32 (2x M-blocked).
__global__ void wmma_conv_kernel(const f16* __restrict__ act, int CSin,
                                 const f16* __restrict__ wpack,
                                 const float* __restrict__ bias,
                                 f16* __restrict__ out, int Cin, int Cout,
                                 int Tin, int Tout, int taps, int stride, int pad) {
  int tTiles = Tout >> 4;
  int tile = blockIdx.x % tTiles;
  int rem  = blockIdx.x / tTiles;
  int n    = rem % NODES;
  int b    = rem / NODES;
  int m0   = blockIdx.y * 32;
  int lane = threadIdx.x & 31;
  int col  = lane & 15;
  int hi   = lane >> 4;
  int Kpad = (Cin + 31) & ~31;
  int t0   = tile << 4;

  v8f acc0, acc1;
#pragma unroll
  for (int r = 0; r < 8; ++r) {
    acc0[r] = bias ? bias[m0 + hi * 8 + r]      : 0.f;
    acc1[r] = bias ? bias[m0 + 16 + hi * 8 + r] : 0.f;
  }

  const size_t rowBase = ((size_t)(b * NODES + n)) * Tin;
  for (int tap = 0; tap < taps; ++tap) {
    int tsrc = (t0 + col) * stride + tap - pad;
    bool tv  = (tsrc >= 0) && (tsrc < Tin);
    int tsc  = tsrc < 0 ? 0 : (tsrc >= Tin ? Tin - 1 : tsrc);  // clamp: always in-bounds
    const f16* bbase = act + (rowBase + tsc) * CSin;
    if (tap + 1 < taps) __builtin_prefetch(bbase + CSin, 0, 0);  // next tap
    const f16* apack = wpack + (size_t)tap * Cout * Kpad;
    for (int kc = 0; kc < Cin; kc += 32) {
      v16h bfrag = tv ? load_fragB(bbase + kc) : zero_frag();
      v16h af0 = load_fragA(apack, m0,      Kpad, kc);
      v16h af1 = load_fragA(apack, m0 + 16, Kpad, kc);
      acc0 = __builtin_amdgcn_wmma_f32_16x16x32_f16(false, af0, false, bfrag,
                                                    (short)0, acc0, false, false);
      acc1 = __builtin_amdgcn_wmma_f32_16x16x32_f16(false, af1, false, bfrag,
                                                    (short)0, acc1, false, false);
    }
  }
  // channels-last output, 8 consecutive channels -> one 16B store per tile
  size_t obase = (((size_t)(b * NODES + n)) * Tout + t0 + col) * Cout + m0 + hi * 8;
  v8h ov0, ov1;
#pragma unroll
  for (int r = 0; r < 8; ++r) { ov0[r] = (f16)acc0[r]; ov1[r] = (f16)acc1[r]; }
  *(v8h*)(out + obase)      = ov0;
  *(v8h*)(out + obase + 16) = ov1;
}

// ---------------- InstanceNorm: partial stats -> finalize -> normalize ----------------
__global__ void in_stats_partial(const f16* __restrict__ x, float* __restrict__ part,
                                 int C, int NT) {
  __shared__ float s1[256], s2[256];
  int s = blockIdx.x % SLICES;
  int b = blockIdx.x / SLICES;
  int L = NT / SLICES;
  size_t base = ((size_t)b * NT + (size_t)s * L) * C;
  float a = 0.f, q = 0.f;
  size_t n = (size_t)L * C;
  for (size_t i = threadIdx.x; i < n; i += 256) {  // coalesced; thread channel = tid % C
    float v = (float)x[base + i];
    a += v; q += v * v;
  }
  s1[threadIdx.x] = a; s2[threadIdx.x] = q;
  __syncthreads();
  for (int off = 128; off >= C; off >>= 1) {       // C is a power of two <= 128
    if ((int)threadIdx.x < off) {
      s1[threadIdx.x] += s1[threadIdx.x + off];
      s2[threadIdx.x] += s2[threadIdx.x + off];
    }
    __syncthreads();
  }
  if ((int)threadIdx.x < C) {
    size_t o = (((size_t)b * C + threadIdx.x) * SLICES + s) * 2;
    part[o + 0] = s1[threadIdx.x];
    part[o + 1] = s2[threadIdx.x];
  }
}

__global__ void in_stats_final(const float* __restrict__ part, float* __restrict__ stats,
                               int C, int NT) {
  int b = blockIdx.x, c = threadIdx.x;
  const float* p = part + (((size_t)b * C + c) * SLICES) * 2;
  float a = 0.f, q = 0.f;
  for (int s = 0; s < SLICES; ++s) { a += p[s * 2]; q += p[s * 2 + 1]; }
  float mean = a / (float)NT;
  float var  = q / (float)NT - mean * mean;
  stats[((size_t)b * C + c) * 2 + 0] = mean;
  stats[((size_t)b * C + c) * 2 + 1] = rsqrtf(var + 1e-5f);
}

// y = relu((x-mean)*rstd) [ + res -> relu again ], vectorized 8 f16 / thread
__global__ void in_norm_relu8(const f16* __restrict__ x, const f16* __restrict__ res,
                              const float* __restrict__ stats, f16* __restrict__ y,
                              int C, int NTC, size_t nvec) {
  size_t v = (size_t)blockIdx.x * 256 + threadIdx.x;
  if (v >= nvec) return;
  size_t i = v * 8;
  int c0 = (int)(i % (size_t)C);          // C multiple of 8 -> 8 elems share c0..c0+7
  int b  = (int)(i / (size_t)NTC);
  const float* st = stats + ((size_t)b * C + c0) * 2;
  v8h xv = *(const v8h*)(x + i);
  v8h rv;
#pragma unroll
  for (int j = 0; j < 8; ++j) rv[j] = (f16)0.f;
  bool hr = (res != nullptr);
  if (hr) rv = *(const v8h*)(res + i);
  v8h ov;
#pragma unroll
  for (int j = 0; j < 8; ++j) {
    float mean = st[j * 2], rstd = st[j * 2 + 1];
    float val = ((float)xv[j] - mean) * rstd;
    val = fmaxf(val, 0.f);
    if (hr) val = fmaxf(val + (float)rv[j], 0.f);
    ov[j] = (f16)val;
  }
  *(v8h*)(y + i) = ov;
}

// ---------------- Small prep kernels ----------------
__global__ void pack_anorm(const float* __restrict__ adj, f16* __restrict__ an) {
  __shared__ float dinv[64];
  int m = threadIdx.x;  // 64 threads
  float d = 0.f;
  if (m < NODES) {
    d = 1.0f;  // self loop (adj diagonal is zero)
    for (int n = 0; n < NODES; ++n) d += adj[m * NODES + n];
  }
  dinv[m] = (d > 0.f) ? rsqrtf(fmaxf(d, 1e-5f)) : 0.f;
  __syncthreads();
  for (int k = 0; k < 64; ++k) {
    float v = 0.f;
    if (m < NODES && k < NODES) {
      float ah = adj[m * NODES + k] + ((m == k) ? 1.f : 0.f);
      v = dinv[m] * ah * dinv[k];
    }
    an[m * 64 + k] = (f16)v;
  }
}

__global__ void pack_weights(const float* __restrict__ src, f16* __restrict__ dst,
                             int M, int K, int Mpad, int Kpad, int taps, int mode) {
  int k  = threadIdx.x;      // < Kpad
  int id = blockIdx.x;       // taps * Mpad
  int tap = id / Mpad;
  int m   = id % Mpad;
  float v = 0.f;
  if (m < M && k < K) {
    if (mode == 0)      v = src[k * M + m];                  // gcn_w (Cin,Cout)
    else if (mode == 1) v = src[(m * K + k) * taps + tap];   // conv_w (Co,Ci,1,9)
    else                v = src[m * K + k];                  // res_w (Co,Ci,1,1)
  }
  dst[((size_t)tap * Mpad + m) * Kpad + k] = (f16)v;
}

__global__ void convert_input(const float* __restrict__ x, f16* __restrict__ act) {
  // x (B,N,T,5) f32 -> act [B][N][T][32] f16, zero-padded channels
  size_t idx = (size_t)blockIdx.x * 256 + threadIdx.x;
  const size_t total = (size_t)BATCH * NODES * TFULL * 32;
  if (idx >= total) return;
  int f = (int)(idx & 31);
  size_t r = idx >> 5;
  int t = (int)(r % TFULL); r /= TFULL;
  int n = (int)(r % NODES);
  int b = (int)(r / NODES);
  float v = (f < 5) ? x[(((size_t)b * NODES + n) * TFULL + t) * 5 + f] : 0.f;
  act[idx] = (f16)v;
}

__global__ void copy_adj(const float* __restrict__ adj, float* __restrict__ dst) {
  int i = blockIdx.x * blockDim.x + threadIdx.x;
  if (i < NODES * NODES) dst[i] = adj[i];
}

// ---------------- Head: node mean, attention pool, FC ----------------
__global__ void mean_nodes(const f16* __restrict__ act, float* __restrict__ H) {
  // act [B][62][512][128] -> H [B][512][128]
  int bt = blockIdx.x;
  int b = bt / THALF, t = bt % THALF;
  int c = threadIdx.x;  // 128, coalesced
  const f16* p = act + (((size_t)b * NODES) * THALF + t) * 128 + c;
  float s = 0.f;
  for (int n = 0; n < NODES; ++n) s += (float)p[(size_t)n * THALF * 128];
  H[((size_t)b * THALF + t) * 128 + c] = s * (1.f / 62.f);
}

__global__ void attn_head(const float* __restrict__ H, const float* __restrict__ w1,
                          const float* __restrict__ b1, const float* __restrict__ w2,
                          const float* __restrict__ b2, const float* __restrict__ fcw,
                          const float* __restrict__ fcb, float* __restrict__ out) {
  __shared__ float sw[512];
  __shared__ float red[512];
  __shared__ float feat[128];
  int b = blockIdx.x, t = threadIdx.x;  // 512 threads
  const float* h = H + ((size_t)b * THALF + t) * 128;
  float score = b2[0];
  for (int j = 0; j < 64; ++j) {
    float hv = b1[j];
    for (int c = 0; c < 128; ++c) hv += h[c] * w1[c * 64 + j];
    score += tanhf(hv) * w2[j];
  }
  red[t] = score; __syncthreads();
  for (int s = 256; s > 0; s >>= 1) { if (t < s) red[t] = fmaxf(red[t], red[t + s]); __syncthreads(); }
  float mx = red[0]; __syncthreads();
  float e = __expf(score - mx);
  red[t] = e; __syncthreads();
  for (int s = 256; s > 0; s >>= 1) { if (t < s) red[t] += red[t + s]; __syncthreads(); }
  float denom = red[0];
  sw[t] = e / denom; __syncthreads();
  if (t < 128) {
    float f = 0.f;
    for (int tt = 0; tt < THALF; ++tt) f += sw[tt] * H[((size_t)b * THALF + tt) * 128 + t];
    feat[t] = f;
  }
  __syncthreads();
  if (t < 3) {
    float l = fcb[t];
    for (int c = 0; c < 128; ++c) l += feat[c] * fcw[c * 3 + t];
    out[b * 3 + t] = l;
  }
}

// ---------------- Host orchestration ----------------
extern "C" void kernel_launch(void* const* d_in, const int* in_sizes, int n_in,
                              void* d_out, int out_size, void* d_ws, size_t ws_size,
                              hipStream_t stream) {
  (void)in_sizes; (void)n_in; (void)out_size; (void)ws_size;

  const float* x   = (const float*)d_in[0];
  const float* adj = (const float*)d_in[1];
  const float* gcn_w[3]  = {(const float*)d_in[2],  (const float*)d_in[8],  (const float*)d_in[14]};
  const float* gcn_b[3]  = {(const float*)d_in[3],  (const float*)d_in[9],  (const float*)d_in[15]};
  const float* conv_w[3] = {(const float*)d_in[4],  (const float*)d_in[10], (const float*)d_in[16]};
  const float* conv_b[3] = {(const float*)d_in[5],  (const float*)d_in[11], (const float*)d_in[17]};
  const float* res_w[3]  = {(const float*)d_in[6],  (const float*)d_in[12], (const float*)d_in[18]};
  const float* res_b[3]  = {(const float*)d_in[7],  (const float*)d_in[13], (const float*)d_in[19]};
  const float* attn_w1 = (const float*)d_in[20];
  const float* attn_b1 = (const float*)d_in[21];
  const float* attn_w2 = (const float*)d_in[22];
  const float* attn_b2 = (const float*)d_in[23];
  const float* fc_w    = (const float*)d_in[24];
  const float* fc_b    = (const float*)d_in[25];
  float* out = (float*)d_out;

  const int CH[4]   = {5, 32, 64, 128};
  const int STR[3]  = {1, 2, 1};
  const int TIN[3]  = {TFULL, TFULL, THALF};
  const int TOUT[3] = {TFULL, THALF, THALF};

  // Four f16 ping-pong regions; max tensor = Z block1/2 = 32*64*1024*64 elems.
  const size_t REGION = (size_t)BATCH * 64 * TFULL * 64;  // 134,217,728 elems
  const size_t RB = REGION * sizeof(f16);
  char* ws = (char*)d_ws;
  f16* bufA = (f16*)(ws);           // block input / final output (channels-last)
  f16* bufB = (f16*)(ws + RB);      // Z (nodes-last), then normalized GCN output
  f16* bufC = (f16*)(ws + 2 * RB);  // graph-GEMM out, then conv out (channels-last)
  f16* bufD = (f16*)(ws + 3 * RB);  // residual (channels-last)
  size_t off = 4 * RB;
  auto bump = [&](size_t bytes) -> void* {
    void* p = ws + off;
    off += (bytes + 255) & ~(size_t)255;
    return p;
  };
  f16* anorm = (f16*)bump(64 * 64 * sizeof(f16));
  f16 *gpack[3], *cpack[3], *rpack[3];
  for (int i = 0; i < 3; ++i) {
    int Cin = CH[i], Cout = CH[i + 1];
    int KpadG = (Cin + 31) & ~31;
    int KpadC = (Cout + 31) & ~31;
    gpack[i] = (f16*)bump((size_t)Cout * KpadG * sizeof(f16));
    cpack[i] = (f16*)bump((size_t)9 * Cout * KpadC * sizeof(f16));
    rpack[i] = (f16*)bump((size_t)Cout * KpadG * sizeof(f16));
  }
  float* part  = (float*)bump((size_t)BATCH * 128 * SLICES * 2 * sizeof(float));
  float* stats = (float*)bump((size_t)BATCH * 128 * 2 * sizeof(float));
  float* H     = (float*)bump((size_t)BATCH * THALF * 128 * sizeof(float));

  // Prep
  pack_anorm<<<1, 64, 0, stream>>>(adj, anorm);
  copy_adj<<<(NODES * NODES + 255) / 256, 256, 0, stream>>>(adj, out + BATCH * 3);
  {
    size_t tot = (size_t)BATCH * NODES * TFULL * 32;
    convert_input<<<(unsigned)((tot + 255) / 256), 256, 0, stream>>>(x, bufA);
  }
  for (int i = 0; i < 3; ++i) {
    int Cin = CH[i], Cout = CH[i + 1];
    int KpadG = (Cin + 31) & ~31;
    int KpadC = (Cout + 31) & ~31;
    pack_weights<<<Cout, KpadG, 0, stream>>>(gcn_w[i],  gpack[i], Cout, Cin,  Cout, KpadG, 1, 0);
    pack_weights<<<9 * Cout, KpadC, 0, stream>>>(conv_w[i], cpack[i], Cout, Cout, Cout, KpadC, 9, 1);
    pack_weights<<<Cout, KpadG, 0, stream>>>(res_w[i],  rpack[i], Cout, Cin,  Cout, KpadG, 1, 2);
  }

  // Three ST-GCN blocks.
  for (int i = 0; i < 3; ++i) {
    int Cin = CH[i], Cout = CH[i + 1];
    int Tin = TIN[i], Tout = TOUT[i], s = STR[i];
    int CSin = (Cin + 31) & ~31;  // stored channel stride of block input
    int NTi = NODES * Tin, NTo = NODES * Tout;

    dim3 gFeat((Tin / 16) * NODES * BATCH, Cout / 32);
    wmma_feat_kernel<<<gFeat, 32, 0, stream>>>(bufA, CSin, gpack[i], bufB, Cin, Cout, Tin);

    dim3 gGraph((Tin / 16) * Cout * BATCH, 2);
    wmma_graph_kernel<<<gGraph, 32, 0, stream>>>(bufB, anorm, gcn_b[i], bufC, Cout, Tin);

    // IN + ReLU on GCN output -> bufB (channels-last)
    in_stats_partial<<<BATCH * SLICES, 256, 0, stream>>>(bufC, part, Cout, NTi);
    in_stats_final<<<BATCH, Cout, 0, stream>>>(part, stats, Cout, NTi);
    {
      size_t nvec = (size_t)BATCH * NTi * Cout / 8;
      in_norm_relu8<<<(unsigned)((nvec + 255) / 256), 256, 0, stream>>>(
          bufC, nullptr, stats, bufB, Cout, NTi * Cout, nvec);
    }

    dim3 gConv((Tout / 16) * NODES * BATCH, Cout / 32);
    wmma_conv_kernel<<<gConv, 32, 0, stream>>>(bufB, Cout, cpack[i], conv_b[i], bufC,
                                               Cout, Cout, Tin, Tout, 9, s, 4);
    wmma_conv_kernel<<<gConv, 32, 0, stream>>>(bufA, CSin, rpack[i], res_b[i], bufD,
                                               Cin, Cout, Tin, Tout, 1, s, 0);

    // IN + ReLU + residual add + ReLU -> bufA (next block input)
    in_stats_partial<<<BATCH * SLICES, 256, 0, stream>>>(bufC, part, Cout, NTo);
    in_stats_final<<<BATCH, Cout, 0, stream>>>(part, stats, Cout, NTo);
    {
      size_t nvec = (size_t)BATCH * NTo * Cout / 8;
      in_norm_relu8<<<(unsigned)((nvec + 255) / 256), 256, 0, stream>>>(
          bufC, bufD, stats, bufA, Cout, NTo * Cout, nvec);
    }
  }

  // Head
  mean_nodes<<<BATCH * THALF, 128, 0, stream>>>(bufA, H);
  attn_head<<<BATCH, 512, 0, stream>>>(H, attn_w1, attn_b1, attn_w2, attn_b2, fc_w, fc_b, out);
}
